// AdaptiveClusteringAttention_17197049053472
// MI455X (gfx1250) — matrix operand order repaired and checked by reading, not compiled, not run on hardware
//
#include <hip/hip_runtime.h>
#include <hip/hip_bf16.h>
#include <math.h>

typedef __attribute__((ext_vector_type(16))) _Float16 v16h;
typedef __attribute__((ext_vector_type(8)))  _Float16 v8h;
typedef __attribute__((ext_vector_type(4)))  _Float16 v4h;
typedef __attribute__((ext_vector_type(8)))  float    v8f;
typedef __attribute__((ext_vector_type(4)))  int      v4i;

#define DEV __device__ __forceinline__

static constexpr int Bb   = 8;
static constexpr int Nn   = 4096;
static constexpr int Dd   = 1024;
static constexpr int Hh   = 16;
static constexpr int DHd  = 64;
static constexpr int Cc   = 256;
static constexpr int Mrows = Bb * Nn;   // 32768
static constexpr int N3    = 3 * Dd;    // 3072

// Padded LDS row strides (halves). All are multiples of 8 halves (16B, keeps
// b128 access aligned) and chosen so 16 lanes hit 16 distinct banks:
//   56 halves = 112B = 28 banks (gcd(28,64)=4, period 16 -> conflict free)
//   72 halves = 144B = 36 banks (period 16 -> conflict free)
//  264 halves = 528B = 132 banks ≡ 4 mod 64 -> cols map to banks 4*c
static constexpr int LDS_KS = 56;   // stride for 32-half K tiles (GEMM A/B, pS)
static constexpr int KC_S   = 72;   // stride for kc rows (64 halves data)
static constexpr int VC_S   = 264;  // stride for vcT rows (256 halves data)

DEV v8f wmma_f16f32(v16h a, v16h b, v8f c) {
  // D = A(16x32 f16) * B(32x16 f16) + C(16x16 f32)
  return __builtin_amdgcn_wmma_f32_16x16x32_f16(false, a, false, b, (short)0, c,
                                                false, false);
}

DEV v8f zero8() {
  v8f z;
#pragma unroll
  for (int i = 0; i < 8; ++i) z[i] = 0.f;
  return z;
}

// ----- CDNA5 async global->LDS copy (ASYNCcnt path), with safe fallback -----
DEV void async_cp_b128(const void* g, void* l) {
#if __has_builtin(__builtin_amdgcn_global_load_async_to_lds_b128)
  __builtin_amdgcn_global_load_async_to_lds_b128(
      (__attribute__((address_space(1))) v4i*)(g),
      (__attribute__((address_space(3))) v4i*)(l), 0, 0);
#else
  *(float4*)l = *(const float4*)g;   // synchronous fallback
#endif
}

template <int N>
DEV void wait_async() {
#if __has_builtin(__builtin_amdgcn_s_wait_asynccnt)
  __builtin_amdgcn_s_wait_asynccnt(N);
#else
  asm volatile("s_wait_asynccnt %0" ::"i"(N) : "memory");
#endif
}

// A fragment (16x32, row-major source, ld in halves).
// Lanes 0-15: row=lane, K {0..7,16..23}; lanes 16-31: row=lane-16, K {8..15,24..31}.
DEV v16h frag_a_ld(const _Float16* __restrict__ tile, int ld, int lane) {
  int row  = lane & 15;
  int koff = (lane >> 4) * 8;
  const _Float16* p = tile + (size_t)row * ld + koff;
  v8h lo = *(const v8h*)(p);
  v8h hi = *(const v8h*)(p + 16);
  v16h r;
#pragma unroll
  for (int i = 0; i < 8; ++i) { r[i] = lo[i]; r[i + 8] = hi[i]; }
  return r;
}

// B fragment (32x16) from (N x K) row-major source, ld in halves.
// Lanes 0-15: col=lane, K {0..7,16..23}; lanes 16-31: col=lane-16, K {8..15,24..31}.
DEV v16h frag_b_ld(const _Float16* __restrict__ tileT, int ld, int lane) {
  int col  = lane & 15;
  int koff = (lane >> 4) * 8;
  const _Float16* p = tileT + (size_t)col * ld + koff;
  v8h lo = *(const v8h*)(p);
  v8h hi = *(const v8h*)(p + 16);
  v16h r;
#pragma unroll
  for (int i = 0; i < 8; ++i) { r[i] = lo[i]; r[i + 8] = hi[i]; }
  return r;
}

// ---------------------------------------------------------------------------
__global__ void cvt_f32_f16(const float* __restrict__ src, _Float16* __restrict__ dst, int n4) {
  int i = blockIdx.x * blockDim.x + threadIdx.x;
  int stride = gridDim.x * blockDim.x;
  const float4* s4 = (const float4*)src;
  for (; i < n4; i += stride) {
    float4 v = s4[i];
    v4h h; h[0] = (_Float16)v.x; h[1] = (_Float16)v.y; h[2] = (_Float16)v.z; h[3] = (_Float16)v.w;
    *(v4h*)&dst[(size_t)i * 4] = h;
  }
}

// WT[j][k] = [w_q | w_kv][k][j]  (3d x d, f16)
__global__ void build_wt(const float* __restrict__ w_q, const float* __restrict__ w_kv,
                         _Float16* __restrict__ WT) {
  size_t id = (size_t)blockIdx.x * blockDim.x + threadIdx.x;
  int j = (int)(id >> 10);
  int k = (int)(id & 1023);
  float v = (j < Dd) ? w_q[(size_t)k * Dd + j] : w_kv[(size_t)k * (2 * Dd) + (j - Dd)];
  WT[id] = (_Float16)v;
}

__global__ void build_wpt(const float* __restrict__ w_proj, _Float16* __restrict__ WPT) {
  size_t id = (size_t)blockIdx.x * blockDim.x + threadIdx.x;
  int j = (int)(id >> 10);
  int k = (int)(id & 1023);
  WPT[id] = (_Float16)w_proj[(size_t)k * Dd + j];
}

__global__ void zero_f32(float* __restrict__ p, int n) {
  int i = blockIdx.x * blockDim.x + threadIdx.x;
  int stride = gridDim.x * blockDim.x;
  for (; i < n; i += stride) p[i] = 0.f;
}

// ---------------------------------------------------------------------------
// Double-buffered async WMMA GEMM: C(MxN) = A(MxK) * BT(NxK)^T, f16 in, K%32==0.
// Block: 256 threads (8 waves), tile 128x128, K-step 32.
// mode 0: f16 out; mode 1: f32 + bias.
// ---------------------------------------------------------------------------
__global__ __launch_bounds__(256) void gemm_wmma(
    const _Float16* __restrict__ A, const _Float16* __restrict__ BT,
    _Float16* __restrict__ Cf16, float* __restrict__ Cf32,
    const float* __restrict__ bias, int M, int N, int K, int mode) {
  __shared__ __align__(32) _Float16 As[2][128 * LDS_KS];
  __shared__ __align__(32) _Float16 Bs[2][128 * LDS_KS];

  int tid  = threadIdx.x;
  int lane = tid & 31;
  int w    = tid >> 5;
  int wm   = (w & 1) * 64;
  int wn   = (w >> 1) * 32;
  int m0   = blockIdx.y * 128;
  int n0   = blockIdx.x * 128;

  v8f acc[4][2];
#pragma unroll
  for (int i = 0; i < 4; ++i)
#pragma unroll
    for (int j = 0; j < 2; ++j) acc[i][j] = zero8();

  int lr = tid >> 1;              // 0..127 (row within tile)
  int lk = (tid & 1) * 16;        // 0 or 16 (K-half chunk)

  const _Float16* agBase = A  + (size_t)(m0 + lr) * K + lk;
  const _Float16* bgBase = BT + (size_t)(n0 + lr) * K + lk;

  auto issue = [&](int s, int buf) {
    const _Float16* ag = agBase + s * 32;
    const _Float16* bg = bgBase + s * 32;
    _Float16* la = &As[buf][lr * LDS_KS + lk];
    _Float16* lb = &Bs[buf][lr * LDS_KS + lk];
    async_cp_b128(ag,     la);
    async_cp_b128(ag + 8, la + 8);
    async_cp_b128(bg,     lb);
    async_cp_b128(bg + 8, lb + 8);
  };

  int stages = K / 32;
  issue(0, 0);
  for (int s = 0; s < stages; ++s) {
    int buf = s & 1;
    if (s + 1 < stages) {
      issue(s + 1, buf ^ 1);   // prefetch next stage into the other buffer
      wait_async<4>();         // current stage (issued earlier) has landed
    } else {
      wait_async<0>();
    }
    __syncthreads();

    v16h bf[2];
#pragma unroll
    for (int j = 0; j < 2; ++j)
      bf[j] = frag_b_ld(&Bs[buf][(wn + j * 16) * LDS_KS], LDS_KS, lane);
#pragma unroll
    for (int i = 0; i < 4; ++i) {
      v16h af = frag_a_ld(&As[buf][(wm + i * 16) * LDS_KS], LDS_KS, lane);
#pragma unroll
      for (int j = 0; j < 2; ++j) acc[i][j] = wmma_f16f32(af, bf[j], acc[i][j]);
    }
    __syncthreads();
  }

  int colb = lane & 15;
  int rsel = (lane >> 4) * 8;
#pragma unroll
  for (int i = 0; i < 4; ++i) {
#pragma unroll
    for (int j = 0; j < 2; ++j) {
      int row = m0 + wm + i * 16 + rsel;
      int col = n0 + wn + j * 16 + colb;
      if (mode == 0) {
#pragma unroll
        for (int r = 0; r < 8; ++r)
          Cf16[(size_t)(row + r) * N + col] = (_Float16)acc[i][j][r];
      } else {
        float bv = bias[col];
#pragma unroll
        for (int r = 0; r < 8; ++r)
          Cf32[(size_t)(row + r) * N + col] = acc[i][j][r] + bv;
      }
    }
  }
}

// ---------------------------------------------------------------------------
// Cluster center accumulation: cent(b,C,2d) f32 sums via native f32 atomics.
// ---------------------------------------------------------------------------
__global__ __launch_bounds__(256) void center_accum(
    const int* __restrict__ cluster, const _Float16* __restrict__ qkv,
    float* __restrict__ cent, int* __restrict__ cnt) {
  int b   = blockIdx.y;
  int n0  = blockIdx.x * 128;
  int tid = threadIdx.x;
  for (int i = 0; i < 128; ++i) {
    int tok = n0 + i;
    int c   = cluster[(size_t)b * Nn + tok];
    const _Float16* src = qkv + ((size_t)b * Nn + tok) * N3 + Dd + tid * 8;
    float* dst = cent + ((size_t)b * Cc + c) * (2 * Dd) + tid * 8;
    v8h v = *(const v8h*)src;
#pragma unroll
    for (int j = 0; j < 8; ++j)
      __hip_atomic_fetch_add(&dst[j], (float)v[j], __ATOMIC_RELAXED, __HIP_MEMORY_SCOPE_AGENT);
    if (tid == 0) atomicAdd(&cnt[(size_t)b * Cc + c], 1);
  }
}

// Normalize sums -> kc (b,h,C,dh) f16, vcT (b,h,dh,C) f16; logw = log(count)|-inf
__global__ __launch_bounds__(256) void center_norm(
    const float* __restrict__ cent, const int* __restrict__ cnt,
    _Float16* __restrict__ kc, _Float16* __restrict__ vct, float* __restrict__ logw) {
  int bc = blockIdx.x;
  int b  = bc >> 8;
  int c  = bc & 255;
  int n  = cnt[bc];
  float inv = (n > 0) ? 1.f / (float)n : 0.f;
  if (threadIdx.x == 0) logw[bc] = (n > 0) ? __logf((float)n) : -INFINITY;
  const float* src = cent + (size_t)bc * (2 * Dd);
  int j0 = threadIdx.x * 8;
#pragma unroll
  for (int jj = 0; jj < 8; ++jj) {
    int j = j0 + jj;
    float v = src[j] * inv;
    if (j < Dd) {
      int h = j >> 6, dhi = j & 63;
      kc[(((size_t)(b * Hh + h)) * Cc + c) * DHd + dhi] = (_Float16)v;
    } else {
      int j2 = j - Dd;
      int h = j2 >> 6, dhi = j2 & 63;
      vct[(((size_t)(b * Hh + h)) * DHd + dhi) * Cc + c] = (_Float16)v;
    }
  }
}

// ---------------------------------------------------------------------------
// Flash attention over 256 cluster centers.
// Block: 256 threads (8 waves), each wave a 16-row tile; grid (n/128, h, b).
// kc (C x dh) and vcT (dh x C) are async-staged once per block into padded LDS.
// score = (q.kc)/8 + log(count); online softmax; out = attn @ vc
// ---------------------------------------------------------------------------
__global__ __launch_bounds__(256) void attn_centers(
    const _Float16* __restrict__ qkv, const _Float16* __restrict__ kc,
    const _Float16* __restrict__ vct, const float* __restrict__ logw,
    _Float16* __restrict__ att) {
  __shared__ __align__(32) _Float16 sKc[Cc * KC_S];        // 256 rows x 64 (pad 72)
  __shared__ __align__(32) _Float16 sVc[DHd * VC_S];       // 64 rows x 256 (pad 264)
  __shared__ __align__(32) _Float16 pS[8][16 * LDS_KS];    // per-wave prob tile

  int tid  = threadIdx.x;
  int lane = tid & 31;
  int w    = tid >> 5;
  int b    = blockIdx.z;
  int h    = blockIdx.y;
  int n0   = blockIdx.x * 128 + w * 16;

  const _Float16* kcb = kc  + (size_t)(b * Hh + h) * Cc * DHd;   // (C, dh)
  const _Float16* vcb = vct + (size_t)(b * Hh + h) * DHd * Cc;   // (dh, C)

  // stage kc: one 64-half row per thread (8 x b128)
  {
    const _Float16* src = kcb + (size_t)tid * DHd;
    _Float16* dst = &sKc[tid * KC_S];
#pragma unroll
    for (int j = 0; j < 8; ++j) async_cp_b128(src + j * 8, dst + j * 8);
  }
  // stage vcT: quarter row (64 halves) per thread
  {
    int row = tid >> 2, part = tid & 3;
    const _Float16* src = vcb + (size_t)row * Cc + part * 64;
    _Float16* dst = &sVc[row * VC_S + part * 64];
#pragma unroll
    for (int j = 0; j < 8; ++j) async_cp_b128(src + j * 8, dst + j * 8);
  }

  const _Float16* qbase = qkv + ((size_t)b * Nn + n0) * N3 + h * DHd;
  v16h qa0 = frag_a_ld(qbase,      N3, lane);   // dh 0..31
  v16h qa1 = frag_a_ld(qbase + 32, N3, lane);   // dh 32..63
  const float* lw = logw + (size_t)b * Cc;

  wait_async<0>();
  __syncthreads();

  float Mr[8], Lr[8];
#pragma unroll
  for (int r = 0; r < 8; ++r) { Mr[r] = -INFINITY; Lr[r] = 0.f; }
  v8f O[4];
#pragma unroll
  for (int t = 0; t < 4; ++t) O[t] = zero8();

  _Float16* ps = pS[w];
  int colb = lane & 15;
  int rsel = (lane >> 4) * 8;

  for (int p = 0; p < 8; ++p) {            // 8 x 32 centers
    v8f s[2];
#pragma unroll
    for (int t = 0; t < 2; ++t) {
      int c0 = p * 32 + t * 16;
      v8f sc = zero8();
      sc = wmma_f16f32(qa0, frag_b_ld(&sKc[c0 * KC_S],      KC_S, lane), sc);
      sc = wmma_f16f32(qa1, frag_b_ld(&sKc[c0 * KC_S + 32], KC_S, lane), sc);
      float lwv = lw[c0 + colb];
#pragma unroll
      for (int r = 0; r < 8; ++r) sc[r] = sc[r] * 0.125f + lwv;
      s[t] = sc;
    }
    float alpha[8];
#pragma unroll
    for (int r = 0; r < 8; ++r) {
      float tm = fmaxf(s[0][r], s[1][r]);
#pragma unroll
      for (int off = 1; off < 16; off <<= 1) tm = fmaxf(tm, __shfl_xor(tm, off, 16));
      float nm = fmaxf(fmaxf(Mr[r], tm), -1e30f);
      alpha[r] = __expf(Mr[r] - nm);
      Mr[r] = nm;
      float p0 = __expf(s[0][r] - nm);
      float p1 = __expf(s[1][r] - nm);
      s[0][r] = p0; s[1][r] = p1;
      float rs = p0 + p1;
#pragma unroll
      for (int off = 1; off < 16; off <<= 1) rs += __shfl_xor(rs, off, 16);
      Lr[r] = Lr[r] * alpha[r] + rs;
    }
#pragma unroll
    for (int t = 0; t < 4; ++t)
#pragma unroll
      for (int r = 0; r < 8; ++r) O[t][r] *= alpha[r];

    // transpose prob tile (C-layout -> row-major 16x32) via per-wave LDS
#pragma unroll
    for (int t = 0; t < 2; ++t)
#pragma unroll
      for (int r = 0; r < 8; ++r)
        ps[(r + rsel) * LDS_KS + t * 16 + colb] = (_Float16)s[t][r];
    asm volatile("s_wait_dscnt 0" ::: "memory");
    v16h pa = frag_a_ld(ps, LDS_KS, lane);
#pragma unroll
    for (int t = 0; t < 4; ++t)
      O[t] = wmma_f16f32(pa, frag_b_ld(&sVc[(t * 16) * VC_S + p * 32], VC_S, lane), O[t]);
  }

#pragma unroll
  for (int r = 0; r < 8; ++r) {
    float inv = (Lr[r] > 0.f) ? 1.f / Lr[r] : 0.f;
    int row = n0 + r + rsel;
#pragma unroll
    for (int t = 0; t < 4; ++t)
      att[((size_t)b * Nn + row) * Dd + h * DHd + t * 16 + colb] = (_Float16)(O[t][r] * inv);
  }
}

// ---------------------------------------------------------------------------
extern "C" void kernel_launch(void* const* d_in, const int* in_sizes, int n_in,
                              void* d_out, int out_size, void* d_ws, size_t ws_size,
                              hipStream_t stream) {
  const int*   cluster = (const int*)d_in[0];
  const float* x       = (const float*)d_in[1];
  const float* w_q     = (const float*)d_in[2];
  const float* w_kv    = (const float*)d_in[3];
  const float* w_proj  = (const float*)d_in[4];
  const float* b_proj  = (const float*)d_in[5];
  float* out = (float*)d_out;

  char* ws = (char*)d_ws;
  size_t off = 0;
  auto alloc = [&](size_t bytes) -> void* {
    void* p = (void*)(ws + off);
    off += (bytes + 255) & ~(size_t)255;
    return p;
  };

  _Float16* xh   = (_Float16*)alloc((size_t)Mrows * Dd * 2);
  _Float16* qkv  = (_Float16*)alloc((size_t)Mrows * N3 * 2);
  _Float16* attb = (_Float16*)alloc((size_t)Mrows * Dd * 2);
  _Float16* WT   = (_Float16*)alloc((size_t)N3 * Dd * 2);
  _Float16* WPT  = (_Float16*)alloc((size_t)Dd * Dd * 2);
  float*    cent = (float*)alloc((size_t)Bb * Cc * 2 * Dd * 4);
  int*      cnt  = (int*)alloc((size_t)Bb * Cc * 4);
  float*    logw = (float*)alloc((size_t)Bb * Cc * 4);
  _Float16* kc   = (_Float16*)alloc((size_t)Bb * Hh * Cc * DHd * 2);
  _Float16* vct  = (_Float16*)alloc((size_t)Bb * Hh * DHd * Cc * 2);

  cvt_f32_f16<<<2048, 256, 0, stream>>>(x, xh, Mrows * Dd / 4);
  build_wt<<<(N3 * Dd) / 256, 256, 0, stream>>>(w_q, w_kv, WT);
  build_wpt<<<(Dd * Dd) / 256, 256, 0, stream>>>(w_proj, WPT);
  zero_f32<<<2048, 256, 0, stream>>>(cent, Bb * Cc * 2 * Dd);
  zero_f32<<<8, 256, 0, stream>>>((float*)cnt, Bb * Cc);

  {  // fused QKV projection: (32768 x 1024) x (1024 x 3072)
    dim3 g(N3 / 128, Mrows / 128);
    gemm_wmma<<<g, 256, 0, stream>>>(xh, WT, qkv, nullptr, nullptr,
                                     Mrows, N3, Dd, /*mode=*/0);
  }
  {  // cluster centers
    dim3 g(Nn / 128, Bb);
    center_accum<<<g, 256, 0, stream>>>(cluster, qkv, cent, cnt);
  }
  center_norm<<<Bb * Cc, 256, 0, stream>>>(cent, cnt, kc, vct, logw);
  {  // attention over centers
    dim3 g(Nn / 128, Hh, Bb);
    attn_centers<<<g, 256, 0, stream>>>(qkv, kc, vct, logw, attb);
  }
  {  // output projection + bias
    dim3 g(Dd / 128, Mrows / 128);
    gemm_wmma<<<g, 256, 0, stream>>>(attb, WPT, nullptr, out, b_proj,
                                     Mrows, Dd, Dd, /*mode=*/1);
  }
}
